// DilatedKnnGraph_7937099563611
// MI455X (gfx1250) — matrix operand order repaired and checked by reading, not compiled, not run on hardware
//
#include <hip/hip_runtime.h>
#include <hip/hip_bf16.h>
#include <math.h>

// Problem constants (from reference)
#define BB 8
#define NN 2048
#define DD 512
#define KFULL 18        // K * DILATION
#define KOUT 9          // edges kept per node after dilation
#define NEDGE (BB * NN * KOUT)      // 147456 per edge_index row
#define NATTR (BB * NN * KFULL)     // 294912 edge_attr values

typedef __attribute__((ext_vector_type(16))) _Float16 v16h;
typedef __attribute__((ext_vector_type(8)))  float    v8f;

// ---------------------------------------------------------------------------
// Kernel 1: one pass over x -> row norms + hi/lo f16 split (Markidis).
// One wave per row; lanes stride columns (coalesced).
// ---------------------------------------------------------------------------
__global__ __launch_bounds__(256) void knn_prep_kernel(const float* __restrict__ x,
                                                       float* __restrict__ sq,
                                                       _Float16* __restrict__ xhi,
                                                       _Float16* __restrict__ xlo) {
    const int tid  = threadIdx.x;
    const int lane = tid & 31;
    const int row  = blockIdx.x * 8 + (tid >> 5);
    const size_t base = (size_t)row * DD;
    float s = 0.f;
    for (int k = lane; k < DD; k += 32) {
        float v = x[base + k];
        s += v * v;
        _Float16 h = (_Float16)v;
        _Float16 l = (_Float16)(v - (float)h);
        xhi[base + k] = h;
        xlo[base + k] = l;
    }
    for (int off = 16; off > 0; off >>= 1)
        s += __shfl_xor(s, off, 32);
    if (lane == 0) sq[row] = s;
}

// ---------------------------------------------------------------------------
// Kernel 2: fused Gram (split-f16 WMMA) + running top-18 per row.
// One wave32 per 16-row strip. A strip (hi/lo f16) cached raw in LDS;
// B tiles loaded straight from the pre-split arrays as 32B vectors.
// ---------------------------------------------------------------------------
__global__ __launch_bounds__(32) void knn_topk_kernel(const _Float16* __restrict__ xhi,
                                                      const _Float16* __restrict__ xlo,
                                                      const float* __restrict__ sq,
                                                      float* __restrict__ out) {
    __shared__ _Float16 Ahi[16][DD];   // 16 KB raw rows
    __shared__ _Float16 Alo[16][DD];   // 16 KB
    __shared__ float    sc[16][16];
    __shared__ float    sqrow[16];

    const int lane = threadIdx.x;          // 0..31
    const int ml   = lane & 15;            // A row-in-strip / B col-in-tile
    const int kh   = lane >> 4;            // lane half -> K interleave select

    const int blk        = blockIdx.x;     // 0..1023
    const int row_base   = blk * 16;
    const int b          = row_base / NN;
    const int batch_base = b * NN;

    // ---- Preload A strip (16 contiguous rows = 16 KB hi + 16 KB lo)
    {
        const uint4* sh = (const uint4*)(xhi + (size_t)row_base * DD);
        const uint4* sl = (const uint4*)(xlo + (size_t)row_base * DD);
        uint4* dh = (uint4*)&Ahi[0][0];
        uint4* dl = (uint4*)&Alo[0][0];
        for (int i = lane; i < (16 * DD * 2) / 16; i += 32) {  // 1024 uint4 each
            dh[i] = sh[i];
            dl[i] = sl[i];
        }
        if (lane < 16) sqrow[lane] = sq[row_base + lane];
    }
    __syncthreads();

    // ---- Running top-18 (sorted desc; ties -> smaller index) in registers
    float tv[KFULL];
    int   ti[KFULL];
#pragma unroll
    for (int i = 0; i < KFULL; ++i) { tv[i] = -INFINITY; ti[i] = 0x7fffffff; }

    // ---- Sweep all 128 column tiles of this batch
    for (int t = 0; t < NN / 16; ++t) {
        const int ncol = batch_base + t * 16 + ml;     // this lane's B column
        const _Float16* bh = xhi + (size_t)ncol * DD;
        const _Float16* bl = xlo + (size_t)ncol * DD;
        const float sqc = sq[ncol];

        // prefetch next tile's B rows (global_prefetch_b8)
        if (t + 1 < NN / 16) {
            const size_t nxt = (size_t)(ncol + 16) * DD;
            __builtin_prefetch(xhi + nxt, 0, 1);
            __builtin_prefetch(xlo + nxt, 0, 1);
        }

        v8f acc = {};
#pragma unroll 4
        for (int c = 0; c < 16; ++c) {
            // B operand layout (32x16 f16): lane half kh holds K = c*32+kh*16+e,
            // i.e. 16 contiguous f16 = one aligned 32-byte vector load.
            v16h bhi = *(const v16h*)(bh + c * 32 + kh * 16);
            v16h blo = *(const v16h*)(bl + c * 32 + kh * 16);

            // A operand layout (16x32 f16): lane half kh holds 16 contiguous f16:
            //   K = c*32 + kh*16 + [0..16)
            union { v16h v; uint4 q[2]; } ah, al;
            ah.q[0] = *(const uint4*)&Ahi[ml][c * 32 + kh * 16];
            ah.q[1] = *(const uint4*)&Ahi[ml][c * 32 + kh * 16 + 8];
            al.q[0] = *(const uint4*)&Alo[ml][c * 32 + kh * 16];
            al.q[1] = *(const uint4*)&Alo[ml][c * 32 + kh * 16 + 8];

            // split-f16 product: hi*hi + hi*lo + lo*hi (f32 accumulate)
            acc = __builtin_amdgcn_wmma_f32_16x16x32_f16(false, ah.v, false, bhi,
                                                         (short)0, acc, false, false);
            acc = __builtin_amdgcn_wmma_f32_16x16x32_f16(false, ah.v, false, blo,
                                                         (short)0, acc, false, false);
            acc = __builtin_amdgcn_wmma_f32_16x16x32_f16(false, al.v, false, bhi,
                                                         (short)0, acc, false, false);
        }

        // C layout: VGPR v, lanes 0-15 -> M=v ; lanes 16-31 -> M=v+8 ; N=lane&15
#pragma unroll
        for (int v = 0; v < 8; ++v) {
            const int M = v + 8 * kh;
            sc[M][ml] = 2.0f * acc[v] - sqrow[M] - sqc;
        }
        __syncthreads();

        // lanes 0-15: merge the tile's 16 candidates of their row into top-18
        if (lane < 16) {
            const int idx_base = batch_base + t * 16;
            for (int j = 0; j < 16; ++j) {
                const float v  = sc[lane][j];
                const int   id = idx_base + j;
                bool take = (v > tv[KFULL - 1]) ||
                            (v == tv[KFULL - 1] && id < ti[KFULL - 1]);
                if (take) {
                    tv[KFULL - 1] = v;
                    ti[KFULL - 1] = id;
#pragma unroll
                    for (int p = KFULL - 1; p > 0; --p) {
                        bool sw = (tv[p] > tv[p - 1]) ||
                                  (tv[p] == tv[p - 1] && ti[p] < ti[p - 1]);
                        if (sw) {
                            float fv = tv[p]; tv[p] = tv[p - 1]; tv[p - 1] = fv;
                            int   ii = ti[p]; ti[p] = ti[p - 1]; ti[p - 1] = ii;
                        }
                    }
                }
            }
        }
        __syncthreads();
    }

    // ---- Emit outputs. d_out layout (all float32):
    //   [0, NEDGE)            edge_index row0 = dilated neighbor global ids
    //   [NEDGE, 2*NEDGE)      edge_index row1 = center global ids
    //   [2*NEDGE, +NATTR)     edge_attr  = all 18 neg-dist values per node
    if (lane < 16) {
        const int g = row_base + lane;
#pragma unroll
        for (int tt = 0; tt < KOUT; ++tt) {
            out[(size_t)g * KOUT + tt]         = (float)ti[2 * tt];
            out[(size_t)NEDGE + g * KOUT + tt] = (float)g;
        }
#pragma unroll
        for (int k = 0; k < KFULL; ++k) {
            out[(size_t)2 * NEDGE + (size_t)g * KFULL + k] = tv[k];
        }
    }
}

// ---------------------------------------------------------------------------
extern "C" void kernel_launch(void* const* d_in, const int* in_sizes, int n_in,
                              void* d_out, int out_size, void* d_ws, size_t ws_size,
                              hipStream_t stream) {
    (void)in_sizes; (void)n_in; (void)out_size; (void)ws_size;
    const float* x   = (const float*)d_in[0];
    float*       out = (float*)d_out;

    // Workspace layout: sq (64 KB) | xhi (16 MB) | xlo (16 MB)
    float*    sq  = (float*)d_ws;
    _Float16* xhi = (_Float16*)((char*)d_ws + 64 * 1024);
    _Float16* xlo = xhi + (size_t)BB * NN * DD;

    // 1) one pass: row norms + hi/lo f16 split
    knn_prep_kernel<<<(BB * NN) / 8, 256, 0, stream>>>(x, sq, xhi, xlo);

    // 2) fused split-f16 WMMA Gram + top-18: one wave per 16-row strip
    knn_topk_kernel<<<(BB * NN) / 16, 32, 0, stream>>>(xhi, xlo, sq, out);
}
